// RPN_48404281426049
// MI455X (gfx1250) — matrix operand order repaired and checked by reading, not compiled, not run on hardware
//
#include <hip/hip_runtime.h>
#include <cstdint>

typedef __bf16 v16bf __attribute__((ext_vector_type(16)));
typedef float  v8f   __attribute__((ext_vector_type(8)));
typedef float  v4f   __attribute__((ext_vector_type(4)));

#define CIN 256
#define NIMG 2
#define NCAT 4768
#define POSTK 1000
#define SCALE_CLAMP 4.135166556742356f

// ---------------------------------------------------------------------------
// Kernel 0: zero the 512B border row used by out-of-bounds conv lanes.
// ---------------------------------------------------------------------------
__global__ __launch_bounds__(256) void rpn_zero_row(__bf16* __restrict__ zrow)
{
    zrow[threadIdx.x] = (__bf16)0.0f;
}

// ---------------------------------------------------------------------------
// Kernel 1: pack conv weights (256,256,3,3) f32 -> bf16 WMMA A-fragments.
// Fragment f = ((octile*9 + t)*8 + ctile), 32 lanes x 16 bf16 contiguous/lane.
// A packing used here (B reads the same k order): lane m & lane m+16; K(e) = kbase + e (kbase 0/16).
// ---------------------------------------------------------------------------
__global__ __launch_bounds__(256) void rpn_pack_weights(
    const float* __restrict__ conv_w, __bf16* __restrict__ wpk)
{
    int g = blockIdx.x * 256 + threadIdx.x;
    if (g >= 256 * 9 * 256) return;
    int e      = g & 15;
    int lane   = (g >> 4) & 31;
    int f      = g >> 9;
    int ct     = f & 7;
    int t      = (f >> 3) % 9;
    int octile = f / 72;
    int oc     = octile * 16 + (lane & 15);
    int kbase  = (lane >= 16) ? 16 : 0;
    int K      = kbase + e;
    int c      = ct * 32 + K;
    wpk[g] = (__bf16)conv_w[((size_t)oc * CIN + c) * 9 + t];
}

// ---------------------------------------------------------------------------
// Kernel 2: NCHW f32 -> NHWC bf16 tile transpose (16 pixels x 256 ch / block).
// ---------------------------------------------------------------------------
__global__ __launch_bounds__(256) void rpn_to_nhwc(
    const float* __restrict__ x, __bf16* __restrict__ xn, int HW)
{
    __shared__ float lds[16 * 256];          // [pix][ch]
    int img = blockIdx.y;
    int p0  = blockIdx.x * 16;
    int c   = threadIdx.x;                   // one channel per thread
    const float* src = x + ((size_t)img * CIN + c) * HW + p0;
    const v4f* s4 = reinterpret_cast<const v4f*>(src);
    #pragma unroll
    for (int q = 0; q < 4; ++q) {
        v4f v = s4[q];
        #pragma unroll
        for (int e = 0; e < 4; ++e)
            lds[(q * 4 + e) * 256 + c] = v[e];
    }
    __syncthreads();
    int pix = threadIdx.x >> 4;
    int cc  = (threadIdx.x & 15) * 16;
    v16bf o;
    #pragma unroll
    for (int e = 0; e < 16; ++e)
        o[e] = (__bf16)lds[pix * 256 + cc + e];
    *reinterpret_cast<v16bf*>(xn + ((size_t)img * HW + p0 + pix) * 256 + cc) = o;
}

// ---------------------------------------------------------------------------
// Kernel 3: 3x3 conv 256->256 implicit GEMM, bf16 WMMA, fused bias+ReLU.
// Wave tile: 16 outC x 64 pixels (4 accumulators; A reused 4x per load).
// Input NHWC bf16; out-of-bounds taps read a 512B zero row (no divergence,
// EXEC stays all-ones around every WMMA). Output t: [img][pixel][256] bf16.
// ---------------------------------------------------------------------------
__device__ __forceinline__ const __bf16* rpn_row_ptr(
    const __bf16* xbase, const __bf16* zrow,
    int p, int lw, int W, int H, int ky, int kx)
{
    int py = p >> lw, px = p & (W - 1);
    int yy = py + ky - 1, xx = px + kx - 1;
    bool inb = ((unsigned)yy < (unsigned)H) && ((unsigned)xx < (unsigned)W);
    return inb ? (xbase + ((size_t)((yy << lw) + xx) << 8)) : zrow;
}

__global__ __launch_bounds__(256) void rpn_conv3x3_wmma(
    const __bf16* __restrict__ xn, const __bf16* __restrict__ wpk,
    const float* __restrict__ conv_b, const __bf16* __restrict__ zrow,
    __bf16* __restrict__ tout, int H, int W, int lw)
{
    int lane = threadIdx.x & 31;
    int wave = threadIdx.x >> 5;
    int HW = H * W;
    int tile64 = blockIdx.x * 8 + wave;
    if (tile64 >= (HW >> 6)) return;
    int octile = blockIdx.y;
    int img = blockIdx.z;

    int n     = lane & 15;
    int khalf = (lane >= 16) ? 16 : 0;
    int pbase = tile64 * 64;

    v8f acc0 = {0.f,0.f,0.f,0.f,0.f,0.f,0.f,0.f};
    v8f acc1 = acc0, acc2 = acc0, acc3 = acc0;
    const __bf16* xbase = xn + (size_t)img * HW * 256;

    for (int t = 0; t < 9; ++t) {
        int ky = t / 3, kx = t % 3;
        const __bf16* r0 = rpn_row_ptr(xbase, zrow, pbase +  0 + n, lw, W, H, ky, kx) + khalf;
        const __bf16* r1 = rpn_row_ptr(xbase, zrow, pbase + 16 + n, lw, W, H, ky, kx) + khalf;
        const __bf16* r2 = rpn_row_ptr(xbase, zrow, pbase + 32 + n, lw, W, H, ky, kx) + khalf;
        const __bf16* r3 = rpn_row_ptr(xbase, zrow, pbase + 48 + n, lw, W, H, ky, kx) + khalf;
        const __bf16* wb = wpk + (size_t)((octile * 9 + t) * 8) * 512 + lane * 16;
        #pragma unroll
        for (int ct = 0; ct < 8; ++ct) {
            v16bf a  = *reinterpret_cast<const v16bf*>(wb + ct * 512);
            v16bf b0 = *reinterpret_cast<const v16bf*>(r0 + ct * 32);
            acc0 = __builtin_amdgcn_wmma_f32_16x16x32_bf16(false, a, false, b0, (short)0, acc0, false, false);
            v16bf b1 = *reinterpret_cast<const v16bf*>(r1 + ct * 32);
            acc1 = __builtin_amdgcn_wmma_f32_16x16x32_bf16(false, a, false, b1, (short)0, acc1, false, false);
            v16bf b2 = *reinterpret_cast<const v16bf*>(r2 + ct * 32);
            acc2 = __builtin_amdgcn_wmma_f32_16x16x32_bf16(false, a, false, b2, (short)0, acc2, false, false);
            v16bf b3 = *reinterpret_cast<const v16bf*>(r3 + ct * 32);
            acc3 = __builtin_amdgcn_wmma_f32_16x16x32_bf16(false, a, false, b3, (short)0, acc3, false, false);
        }
    }

    // C/D layout: VGPR r: lanes 0-15 -> M=r, lanes 16-31 -> M=8+r; N = lane&15.
    int mh = (lane >= 16) ? 8 : 0;
    float bias[8];
    #pragma unroll
    for (int r = 0; r < 8; ++r) bias[r] = conv_b[octile * 16 + mh + r];

    auto epi = [&](v8f& A, int s) {
        int p = pbase + s * 16 + n;
        __bf16* trow = tout + ((size_t)img * HW + p) * 256;
        #pragma unroll
        for (int r = 0; r < 8; ++r) {
            float v = A[r] + bias[r];
            trow[octile * 16 + mh + r] = (__bf16)(v > 0.f ? v : 0.f);
        }
    };
    epi(acc0, 0); epi(acc1, 1); epi(acc2, 2); epi(acc3, 3);
}

// ---------------------------------------------------------------------------
// Kernel 4: fused 1x1 heads (3 obj + 12 delta) + anchors + apply_deltas + clip.
// ---------------------------------------------------------------------------
__global__ __launch_bounds__(256) void rpn_head_proposals(
    const __bf16* __restrict__ tin,
    const float* __restrict__ obj_w, const float* __restrict__ obj_b,
    const float* __restrict__ delta_w, const float* __restrict__ delta_b,
    float* __restrict__ scores, float* __restrict__ boxes,
    int H, int W, float stride, float asize)
{
    __shared__ float wsm[15 * 256];
    __shared__ float bsm[15];
    for (int i = threadIdx.x; i < 15 * 256; i += 256)
        wsm[i] = (i < 3 * 256) ? obj_w[i] : delta_w[i - 3 * 256];
    if (threadIdx.x < 15)
        bsm[threadIdx.x] = (threadIdx.x < 3) ? obj_b[threadIdx.x]
                                             : delta_b[threadIdx.x - 3];
    __syncthreads();

    int HW = H * W;
    int gid = blockIdx.x * 256 + threadIdx.x;
    if (gid >= NIMG * HW) return;
    int img = gid / HW, p = gid % HW;
    int py = p / W, px = p % W;

    const __bf16* trow = tin + ((size_t)img * HW + p) * 256;
    float acc[15];
    #pragma unroll
    for (int j = 0; j < 15; ++j) acc[j] = bsm[j];
    for (int c = 0; c < 256; ++c) {
        float tv = (float)trow[c];
        #pragma unroll
        for (int j = 0; j < 15; ++j) acc[j] += tv * wsm[j * 256 + c];
    }

    float cx = px * stride, cy = py * stride;
    const float ratios[3] = {0.5f, 1.0f, 2.0f};
    for (int a = 0; a < 3; ++a) {
        float r  = ratios[a];
        float wa = asize * rsqrtf(r);   // sqrt(size^2 / r)
        float ha = wa * r;
        float dx = acc[3 + a * 4 + 0], dy = acc[3 + a * 4 + 1];
        float dw = acc[3 + a * 4 + 2], dh = acc[3 + a * 4 + 3];
        dw = fminf(dw, SCALE_CLAMP);
        dh = fminf(dh, SCALE_CLAMP);
        float pcx = dx * wa + cx, pcy = dy * ha + cy;
        float pw = expf(dw) * wa, ph = expf(dh) * ha;
        float x1 = fminf(fmaxf(pcx - 0.5f * pw, 0.f), 1024.f);
        float y1 = fminf(fmaxf(pcy - 0.5f * ph, 0.f), 1024.f);
        float x2 = fminf(fmaxf(pcx + 0.5f * pw, 0.f), 1024.f);
        float y2 = fminf(fmaxf(pcy + 0.5f * ph, 0.f), 1024.f);
        int o = img * (3 * HW) + p * 3 + a;
        scores[o] = acc[a];
        float* bo = boxes + (size_t)o * 4;
        bo[0] = x1; bo[1] = y1; bo[2] = x2; bo[3] = y2;
    }
}

// ---------------------------------------------------------------------------
// Kernel 5: per (img, level): 22-bit radix-select top-k, bitonic sort (desc),
// greedy NMS on sorted boxes, write into concatenated arrays.
// ---------------------------------------------------------------------------
__global__ __launch_bounds__(1024) void rpn_topk_nms(
    const float* __restrict__ scores, const float* __restrict__ boxes,
    float* __restrict__ cat_scores, float* __restrict__ cat_boxes,
    int LVL, int k, int catoff)
{
    __shared__ unsigned hist[2048];
    __shared__ float ss[1024];
    __shared__ int   si[1024];
    __shared__ float bx[1024][4];
    __shared__ float area[1024];
    __shared__ int   keep[1024];
    __shared__ int s_b1, s_rem1, s_b2, s_rem2, s_nsel, s_neq;

    int tid = threadIdx.x;
    int img = blockIdx.x;
    const float* sc  = scores + (size_t)img * LVL;
    const float* bxg = boxes + (size_t)img * LVL * 4;

    hist[tid * 2] = 0; hist[tid * 2 + 1] = 0;
    __syncthreads();
    for (int i = tid; i < LVL; i += 1024) {
        unsigned u = __float_as_uint(sc[i]);
        unsigned key = (u & 0x80000000u) ? ~u : (u | 0x80000000u);
        atomicAdd(&hist[key >> 21], 1u);
    }
    __syncthreads();
    if (tid == 0) {
        int cum = 0, b1 = 0, above = 0;
        for (int b = 2047; b >= 0; --b) {
            int h = (int)hist[b];
            if (cum + h >= k) { b1 = b; above = cum; break; }
            cum += h;
        }
        s_b1 = b1; s_rem1 = k - above;
    }
    __syncthreads();
    int b1 = s_b1, rem1 = s_rem1;

    hist[tid * 2] = 0; hist[tid * 2 + 1] = 0;
    __syncthreads();
    for (int i = tid; i < LVL; i += 1024) {
        unsigned u = __float_as_uint(sc[i]);
        unsigned key = (u & 0x80000000u) ? ~u : (u | 0x80000000u);
        if ((int)(key >> 21) == b1) atomicAdd(&hist[(key >> 10) & 0x7FFu], 1u);
    }
    __syncthreads();
    if (tid == 0) {
        int cum = 0, b2 = 0, above = 0;
        for (int b = 2047; b >= 0; --b) {
            int h = (int)hist[b];
            if (cum + h >= rem1) { b2 = b; above = cum; break; }
            cum += h;
        }
        s_b2 = b2; s_rem2 = rem1 - above;
        s_nsel = 0; s_neq = 0;
    }
    __syncthreads();
    unsigned thr22 = ((unsigned)b1 << 11) | (unsigned)s_b2;
    int rem2 = s_rem2;

    for (int i = tid; i < LVL; i += 1024) {
        float s = sc[i];
        unsigned u = __float_as_uint(s);
        unsigned key = (u & 0x80000000u) ? ~u : (u | 0x80000000u);
        unsigned top22 = key >> 10;
        int slot = -1;
        if (top22 > thr22) {
            slot = atomicAdd(&s_nsel, 1);
        } else if (top22 == thr22) {
            int e = atomicAdd(&s_neq, 1);
            if (e < rem2) slot = atomicAdd(&s_nsel, 1);
        }
        if (slot >= 0) { ss[slot] = s; si[slot] = i; }
    }
    __syncthreads();
    if (tid >= k) { ss[tid] = -3.0e38f; si[tid] = -1; }

    for (int size = 2; size <= 1024; size <<= 1) {
        for (int str = size >> 1; str > 0; str >>= 1) {
            __syncthreads();
            int i = tid, pp = i ^ str;
            if (pp > i) {
                bool up = ((i & size) == 0);   // descending blocks
                float a = ss[i], b = ss[pp];
                if (up ? (a < b) : (a > b)) {
                    ss[i] = b; ss[pp] = a;
                    int ti = si[i]; si[i] = si[pp]; si[pp] = ti;
                }
            }
        }
    }
    __syncthreads();

    float score_t = ss[tid];
    int idx_t = si[tid];
    if (tid < k && idx_t >= 0) {
        const float* b = bxg + (size_t)idx_t * 4;
        float x1 = b[0], y1 = b[1], x2 = b[2], y2 = b[3];
        bx[tid][0] = x1; bx[tid][1] = y1; bx[tid][2] = x2; bx[tid][3] = y2;
        float w = x2 - x1, h = y2 - y1;
        area[tid] = w * h;
        keep[tid] = (w > 0.f) && (h > 0.f);
    } else {
        bx[tid][0] = 0.f; bx[tid][1] = 0.f; bx[tid][2] = 0.f; bx[tid][3] = 0.f;
        area[tid] = 0.f;
        keep[tid] = 0;
    }

    for (int i = 0; i < k; ++i) {
        __syncthreads();
        if (!keep[i]) continue;                       // uniform branch
        if (tid > i && tid < k && keep[tid]) {
            float xx1 = fmaxf(bx[i][0], bx[tid][0]);
            float yy1 = fmaxf(bx[i][1], bx[tid][1]);
            float xx2 = fminf(bx[i][2], bx[tid][2]);
            float yy2 = fminf(bx[i][3], bx[tid][3]);
            float iw = fmaxf(xx2 - xx1, 0.f), ih = fmaxf(yy2 - yy1, 0.f);
            float inter = iw * ih;
            float iou = inter / fmaxf(area[i] + area[tid] - inter, 1e-9f);
            if (iou > 0.7f) keep[tid] = 0;
        }
    }
    __syncthreads();

    if (tid < k) {
        int o = img * NCAT + catoff + tid;
        cat_scores[o] = keep[tid] ? score_t : -1e9f;
        float* co = cat_boxes + (size_t)o * 4;
        co[0] = bx[tid][0]; co[1] = bx[tid][1];
        co[2] = bx[tid][2]; co[3] = bx[tid][3];
    }
}

// ---------------------------------------------------------------------------
// Kernel 6: final top-POST over 4768 concatenated candidates per image.
// Output layout: boxes [2,1000,4] then scores [2,1000], flat float.
// ---------------------------------------------------------------------------
__global__ __launch_bounds__(1024) void rpn_final_topk(
    const float* __restrict__ cat_scores, const float* __restrict__ cat_boxes,
    float* __restrict__ out)
{
    const int SN = 8192;
    __shared__ unsigned key[SN];
    __shared__ unsigned short pidx[SN];
    int tid = threadIdx.x, img = blockIdx.x;
    const float* sc = cat_scores + (size_t)img * NCAT;

    for (int v = 0; v < SN / 1024; ++v) {
        int i = v * 1024 + tid;
        if (i < NCAT) {
            unsigned u = __float_as_uint(sc[i]);
            key[i] = (u & 0x80000000u) ? ~u : (u | 0x80000000u);
            pidx[i] = (unsigned short)i;
        } else {
            key[i] = 0u;
            pidx[i] = 0;
        }
    }

    for (unsigned k2 = 2; k2 <= (unsigned)SN; k2 <<= 1) {
        for (unsigned j = k2 >> 1; j > 0; j >>= 1) {
            __syncthreads();
            for (int v = 0; v < SN / 1024; ++v) {
                unsigned i = v * 1024 + tid;
                unsigned ixj = i ^ j;
                if (ixj > i) {
                    bool up = ((i & k2) == 0);      // descending
                    unsigned a = key[i], b = key[ixj];
                    if (up ? (a < b) : (a > b)) {
                        key[i] = b; key[ixj] = a;
                        unsigned short t2 = pidx[i];
                        pidx[i] = pidx[ixj]; pidx[ixj] = t2;
                    }
                }
            }
        }
    }
    __syncthreads();

    for (int t = tid; t < POSTK; t += 1024) {
        unsigned kk = key[t];
        unsigned u = (kk & 0x80000000u) ? (kk & 0x7FFFFFFFu) : ~kk;
        float s = __uint_as_float(u);
        int id = pidx[t];
        const float* b = cat_boxes + ((size_t)img * NCAT + id) * 4;
        float* ob = out + ((size_t)img * POSTK + t) * 4;
        ob[0] = b[0]; ob[1] = b[1]; ob[2] = b[2]; ob[3] = b[3];
        out[NIMG * POSTK * 4 + img * POSTK + t] = s;
    }
}

// ---------------------------------------------------------------------------
extern "C" void kernel_launch(void* const* d_in, const int* in_sizes, int n_in,
                              void* d_out, int out_size, void* d_ws, size_t ws_size,
                              hipStream_t stream) {
    (void)in_sizes; (void)n_in; (void)out_size; (void)ws_size;
    const float* conv_w  = (const float*)d_in[5];
    const float* conv_b  = (const float*)d_in[6];
    const float* obj_w   = (const float*)d_in[7];
    const float* obj_b   = (const float*)d_in[8];
    const float* delta_w = (const float*)d_in[9];
    const float* delta_b = (const float*)d_in[10];

    char* base = (char*)d_ws;
    size_t off = 0;
    auto carve = [&](size_t bytes) -> char* {
        char* p = base + off;
        off = (off + bytes + 255) & ~(size_t)255;
        return p;
    };
    __bf16* wpk       = (__bf16*)carve((size_t)256 * 9 * 256 * 2);        // 1.18 MB
    __bf16* zrow      = (__bf16*)carve(512);                              // zero row
    __bf16* xnhwc     = (__bf16*)carve((size_t)NIMG * 65536 * 256 * 2);   // 67 MB
    __bf16* tbuf      = (__bf16*)carve((size_t)NIMG * 65536 * 256 * 2);   // 67 MB
    float*  scores_ws = (float*) carve((size_t)NIMG * 196608 * 4);
    float*  boxes_ws  = (float*) carve((size_t)NIMG * 196608 * 16);
    float*  cat_s     = (float*) carve((size_t)NIMG * NCAT * 4);
    float*  cat_b     = (float*) carve((size_t)NIMG * NCAT * 16);

    rpn_zero_row<<<1, 256, 0, stream>>>(zrow);
    rpn_pack_weights<<<2304, 256, 0, stream>>>(conv_w, wpk);

    const int   Hs[5]      = {256, 128, 64, 32, 16};
    const int   lws[5]     = {8, 7, 6, 5, 4};       // log2(W)
    const float strides[5] = {4.f, 8.f, 16.f, 32.f, 64.f};
    const float sizes[5]   = {32.f, 64.f, 128.f, 256.f, 512.f};
    const int   ks[5]      = {1000, 1000, 1000, 1000, 768};
    const int   catoffs[5] = {0, 1000, 2000, 3000, 4000};

    for (int l = 0; l < 5; ++l) {
        int H = Hs[l], W = Hs[l], HW = H * W;

        dim3 gtr(HW / 16, NIMG, 1);
        rpn_to_nhwc<<<gtr, 256, 0, stream>>>((const float*)d_in[l], xnhwc, HW);

        int ntiles64 = HW / 64;
        dim3 gconv((ntiles64 + 7) / 8, 16, NIMG);
        rpn_conv3x3_wmma<<<gconv, 256, 0, stream>>>(
            xnhwc, wpk, conv_b, zrow, tbuf, H, W, lws[l]);

        int tot = NIMG * HW;
        rpn_head_proposals<<<(tot + 255) / 256, 256, 0, stream>>>(
            tbuf, obj_w, obj_b, delta_w, delta_b,
            scores_ws, boxes_ws, H, W, strides[l], sizes[l]);

        rpn_topk_nms<<<NIMG, 1024, 0, stream>>>(
            scores_ws, boxes_ws, cat_s, cat_b, 3 * HW, ks[l], catoffs[l]);
    }

    rpn_final_topk<<<NIMG, 1024, 0, stream>>>(cat_s, cat_b, (float*)d_out);
}
